// NystromAttention_48318382080380
// MI455X (gfx1250) — matrix operand order repaired and checked by reading, not compiled, not run on hardware
//
#include <hip/hip_runtime.h>
#include <hip/hip_bf16.h>

#define DEVINL __device__ __forceinline__

typedef __attribute__((ext_vector_type(16))) __bf16 v16bf;
typedef __attribute__((ext_vector_type(8)))  __bf16 v8bf;
typedef __attribute__((ext_vector_type(8)))  float  v8f;
typedef __attribute__((ext_vector_type(4)))  float  v4f;

// Problem constants (reference: DIM=512, HEADS=8, LANDMARKS=64, B=2, N=16384)
constexpr int Bb = 2, Nn = 16384, Cc = 512, Hh = 8, HD = 64, Mm = 64;
constexpr int BH = Bb * Hh;                // 16
constexpr int NSPLIT = 16;                 // flash-split per (b,h)
constexpr int NCHUNK = (Nn / NSPLIT) / 64; // 16 chunks of 64 tokens

// ---------------------------------------------------------------------------
// WMMA + fragment-staged LDS helpers (ISA 7.12.2 layouts).
// One fragment = 32 lanes x 32B; operand load = one aligned 32B vector load.
// Within any 8-aligned k-block the lane bits are constant, so 8 consecutive k
// are CONTIGUOUS in the fragment -> staging can use 16B vector stores.
// ---------------------------------------------------------------------------
DEVINL v8f wmma_bf16(v16bf a, v16bf b, v8f c) {
  return __builtin_amdgcn_wmma_f32_16x16x32_bf16(false, a, false, b, (short)0, c,
                                                 false, false);
}
DEVINL int a_idx(int m, int k) {  // A: 16x32 element (m,k)
  int frag = ((m >> 4) << 1) + (k >> 5);
  int lane = (((k >> 3) & 1) << 4) + (m & 15);
  int e    = (((k >> 4) & 1) << 3) + (k & 7);
  return (((frag << 5) + lane) << 4) + e;
}
DEVINL int b_idx(int k, int n) {  // B: 32x16 element (k,n)
  int frag = ((n >> 4) << 1) + (k >> 5);
  int lane = (((k >> 3) & 1) << 4) + (n & 15);
  int e    = (((k >> 4) & 1) << 3) + (k & 7);
  return (((frag << 5) + lane) << 4) + e;
}
DEVINL v16bf load_frag(const __bf16* F, int frag) {
  int lane = threadIdx.x & 31;
  return *(const v16bf*)(F + (((frag << 5) + lane) << 4));
}
DEVINL void store_D_lds(float* M, int ld, int row0, int col0, v8f d) {
  int lane = threadIdx.x & 31;
  int c = lane & 15, g = lane >> 4;
  float* base = M + (row0 + 8 * g) * ld + col0 + c;
#pragma unroll
  for (int r = 0; r < 8; ++r) base[r * ld] = d[r];
}
// 8 f32 -> v8bf
DEVINL v8bf cvt8(const float* __restrict__ g) {
  v4f x0 = *(const v4f*)g;
  v4f x1 = *(const v4f*)(g + 4);
  v8bf o;
  o[0] = (__bf16)x0[0]; o[1] = (__bf16)x0[1];
  o[2] = (__bf16)x0[2]; o[3] = (__bf16)x0[3];
  o[4] = (__bf16)x1[0]; o[5] = (__bf16)x1[1];
  o[6] = (__bf16)x1[2]; o[7] = (__bf16)x1[3];
  return o;
}
DEVINL void copy4096(__bf16* dst, const __bf16* src, int t) {
  for (int i = t; i < 512; i += 128)
    *(v8bf*)&dst[i * 8] = *(const v8bf*)&src[i * 8];
}
// 64x64 = 64x64 @ 64x64 on staged fragments -> f32 row-major. 4 waves.
DEVINL void mm64_frag(float* out, const __bf16* Af, const __bf16* Bf) {
  int wave = threadIdx.x >> 5;
  v16bf a0 = load_frag(Af, wave * 2 + 0);
  v16bf a1 = load_frag(Af, wave * 2 + 1);
#pragma unroll
  for (int ct = 0; ct < 4; ++ct) {
    v8f acc = {};
    acc = wmma_bf16(a0, load_frag(Bf, ct * 2 + 0), acc);
    acc = wmma_bf16(a1, load_frag(Bf, ct * 2 + 1), acc);
    store_D_lds(out, 64, wave * 16, ct * 16, acc);
  }
}

// ---------------------------------------------------------------------------
// Kernel 1/7: fused QKV projection (x tile staged once feeds 24 WMMAs).
// ---------------------------------------------------------------------------
__global__ void __launch_bounds__(128)
gemm_qkv(const float* __restrict__ x, const float* __restrict__ wq,
         const float* __restrict__ bq, const float* __restrict__ wk,
         const float* __restrict__ bk, const float* __restrict__ wv,
         const float* __restrict__ bv, __bf16* __restrict__ qb,
         __bf16* __restrict__ kb, __bf16* __restrict__ vb) {
  __shared__ alignas(32) __bf16 sAf[4096];
  __shared__ alignas(32) __bf16 sBf[3][4096];
  int t = threadIdx.x;
  int wave = t >> 5, lane = t & 31, cl = lane & 15, g = lane >> 4;
  int row0 = blockIdx.x * 64;
  int col0 = blockIdx.y * 64;
  v8f acc[3][4] = {};
  for (int kc = 0; kc < 8; ++kc) {
    int k0 = kc * 64;
    for (int idx = t; idx < 512; idx += 128) {
      int k8 = (idx & 7) << 3, m = idx >> 3;
      *(v8bf*)&sAf[a_idx(m, k8)] = cvt8(&x[(size_t)(row0 + m) * Cc + k0 + k8]);
    }
#pragma unroll
    for (int i = 0; i < 3; ++i) {
      const float* W = (i == 0) ? wq : (i == 1) ? wk : wv;
      for (int idx = t; idx < 512; idx += 128) {
        int k8 = (idx & 7) << 3, n = idx >> 3;
        *(v8bf*)&sBf[i][b_idx(k8, n)] = cvt8(&W[(size_t)(col0 + n) * Cc + k0 + k8]);
      }
    }
    if (kc + 1 < 8)
      __builtin_prefetch(&x[(size_t)(row0 + (t >> 1)) * Cc + k0 + 64 + (t & 1) * 32], 0, 1);
    __syncthreads();
    v16bf a0 = load_frag(sAf, wave * 2 + 0);
    v16bf a1 = load_frag(sAf, wave * 2 + 1);
    v16bf b0 = load_frag(sBf[0], 0), b1 = load_frag(sBf[0], 1);
#pragma unroll
    for (int i = 0; i < 3; ++i)
#pragma unroll
      for (int ct = 0; ct < 4; ++ct) {
        v16bf n0 = b0, n1 = b1;
        if (!(i == 2 && ct == 3)) {  // prefetch next B fragment pair
          int ni = (ct == 3) ? i + 1 : i;
          int nct = (ct == 3) ? 0 : ct + 1;
          n0 = load_frag(sBf[ni], nct * 2 + 0);
          n1 = load_frag(sBf[ni], nct * 2 + 1);
        }
        acc[i][ct] = wmma_bf16(a0, b0, acc[i][ct]);
        acc[i][ct] = wmma_bf16(a1, b1, acc[i][ct]);
        b0 = n0; b1 = n1;
      }
    __syncthreads();
  }
#pragma unroll
  for (int i = 0; i < 3; ++i) {
    const float* bias = (i == 0) ? bq : (i == 1) ? bk : bv;
    __bf16* outp = (i == 0) ? qb : (i == 1) ? kb : vb;
    float scale = (i == 0) ? 0.125f : 1.0f;
#pragma unroll
    for (int ct = 0; ct < 4; ++ct)
#pragma unroll
      for (int r = 0; r < 8; ++r) {
        int gr = row0 + wave * 16 + r + 8 * g;
        int gc = col0 + ct * 16 + cl;
        float v = (acc[i][ct][r] + bias[gc]) * scale;
        int b = gr >> 14, n = gr & (Nn - 1);
        int h = gc >> 6, d = gc & 63;
        outp[(((size_t)(b * Hh + h)) * Nn + n) * HD + d] = (__bf16)v;
      }
  }
}

// ---------------------------------------------------------------------------
// Kernel 2/7: landmark means; stored PRE-STAGED in HBM (qland A, kland^T B).
// ---------------------------------------------------------------------------
__global__ void __launch_bounds__(64)
landmarks(const __bf16* __restrict__ qb, const __bf16* __restrict__ kb,
          __bf16* __restrict__ qlandA, __bf16* __restrict__ klandB) {
  int bh = blockIdx.x >> 6;
  int mi = blockIdx.x & 63;
  int d = threadIdx.x;
  const __bf16* qp = qb + ((size_t)bh * Nn + mi * 256) * HD + d;
  const __bf16* kp = kb + ((size_t)bh * Nn + mi * 256) * HD + d;
  float sq = 0.f, sk = 0.f;
  for (int j = 0; j < 256; ++j) {
    sq += (float)qp[(size_t)j * HD];
    sk += (float)kp[(size_t)j * HD];
  }
  qlandA[(size_t)bh * 4096 + a_idx(mi, d)] = (__bf16)(sq * (1.f / 256.f));
  klandB[(size_t)bh * 4096 + b_idx(d, mi)] = (__bf16)(sk * (1.f / 256.f));
}

// ---------------------------------------------------------------------------
// Kernel 3/7: ker2 softmax + iterative Moore-Penrose pinv (6 Newton-Schulz).
// ---------------------------------------------------------------------------
__global__ void __launch_bounds__(128)
ker2_pinv(const __bf16* __restrict__ qlandA, const __bf16* __restrict__ klandB,
          __bf16* __restrict__ zmat) {
  __shared__ alignas(32) __bf16 sAa[4096];
  __shared__ alignas(32) __bf16 sZa[4096], sZb[4096];
  __shared__ alignas(32) __bf16 sAZa[4096], sTb[4096];
  __shared__ alignas(16) float sF[4096];
  __shared__ float red[64];
  __shared__ float sDenom;
  int bh = blockIdx.x;
  int t = threadIdx.x;
  copy4096(sAZa, qlandA + (size_t)bh * 4096, t);
  copy4096(sTb, klandB + (size_t)bh * 4096, t);
  __syncthreads();
  mm64_frag(sF, sAZa, sTb);  // scores = q_land @ k_land^T
  __syncthreads();
  if (t < 64) {  // row softmax -> A (A layout), 8-wide stores
    float mx = -1e30f;
    for (int j = 0; j < 64; j += 4) {
      v4f v = *(const v4f*)&sF[t * 64 + j];
      mx = fmaxf(mx, fmaxf(fmaxf(v[0], v[1]), fmaxf(v[2], v[3])));
    }
    float sum = 0.f;
    for (int j = 0; j < 64; ++j) sum += __expf(sF[t * 64 + j] - mx);
    float inv = 1.f / sum;
    for (int j8 = 0; j8 < 64; j8 += 8) {
      v8bf p;
#pragma unroll
      for (int i = 0; i < 8; ++i)
        p[i] = (__bf16)(__expf(sF[t * 64 + j8 + i] - mx) * inv);
      *(v8bf*)&sAa[a_idx(t, j8)] = p;
    }
  }
  __syncthreads();
  if (t < 64) {  // column sums (row sums == 1 after softmax)
    float cs = 0.f;
    for (int r = 0; r < 64; ++r) cs += (float)sAa[a_idx(r, t)];
    red[t] = cs;
  }
  __syncthreads();
  if (t == 0) {
    float mx = red[0];
    for (int i = 1; i < 64; ++i) mx = fmaxf(mx, red[i]);
    sDenom = mx;
  }
  __syncthreads();
  float inv = 1.f / sDenom;
  for (int e = t; e < 4096; e += 128) {  // Z0 = A^T / denom (both layouts)
    int r = e >> 6, c = e & 63;
    float v = (float)sAa[a_idx(r, c)] * inv;
    sZa[a_idx(c, r)] = (__bf16)v;
    sZb[b_idx(c, r)] = (__bf16)v;
  }
  for (int it = 0; it < 6; ++it) {
    __syncthreads();
    mm64_frag(sF, sAa, sZb);  // AZ
    __syncthreads();
    for (int idx = t; idx < 512; idx += 128) {  // sAZa: vector over k=c
      int c8 = (idx & 7) << 3, r = idx >> 3;
      v8bf o;
      const float* f = &sF[r * 64 + c8];
#pragma unroll
      for (int i = 0; i < 8; ++i) o[i] = (__bf16)f[i];
      *(v8bf*)&sAZa[a_idx(r, c8)] = o;
    }
    for (int idx = t; idx < 512; idx += 128) {  // T1 = 7I - AZ: vector over k=r
      int c = idx & 63, r0 = (idx >> 6) << 3;
      v8bf o;
#pragma unroll
      for (int i = 0; i < 8; ++i)
        o[i] = (__bf16)((((r0 + i) == c) ? 7.f : 0.f) - sF[(r0 + i) * 64 + c]);
      *(v8bf*)&sTb[b_idx(r0, c)] = o;
    }
    __syncthreads();
    mm64_frag(sF, sAZa, sTb);  // T2 = AZ @ T1
    __syncthreads();
    for (int idx = t; idx < 512; idx += 128) {  // T3 = 15I - T2
      int c = idx & 63, r0 = (idx >> 6) << 3;
      v8bf o;
#pragma unroll
      for (int i = 0; i < 8; ++i)
        o[i] = (__bf16)((((r0 + i) == c) ? 15.f : 0.f) - sF[(r0 + i) * 64 + c]);
      *(v8bf*)&sTb[b_idx(r0, c)] = o;
    }
    __syncthreads();
    mm64_frag(sF, sAZa, sTb);  // T4 = AZ @ T3
    __syncthreads();
    for (int idx = t; idx < 512; idx += 128) {  // T5 = 13I - T4
      int c = idx & 63, r0 = (idx >> 6) << 3;
      v8bf o;
#pragma unroll
      for (int i = 0; i < 8; ++i)
        o[i] = (__bf16)((((r0 + i) == c) ? 13.f : 0.f) - sF[(r0 + i) * 64 + c]);
      *(v8bf*)&sTb[b_idx(r0, c)] = o;
    }
    __syncthreads();
    mm64_frag(sF, sZa, sTb);  // Z @ T5
    __syncthreads();
    for (int idx = t; idx < 512; idx += 128) {  // Z (A layout): vector over k=c
      int c8 = (idx & 7) << 3, r = idx >> 3;
      v8bf o;
#pragma unroll
      for (int i = 0; i < 8; ++i) o[i] = (__bf16)(0.25f * sF[r * 64 + c8 + i]);
      *(v8bf*)&sZa[a_idx(r, c8)] = o;
    }
    for (int idx = t; idx < 512; idx += 128) {  // Z (B layout): vector over k=r
      int c = idx & 63, r0 = (idx >> 6) << 3;
      v8bf o;
#pragma unroll
      for (int i = 0; i < 8; ++i) o[i] = (__bf16)(0.25f * sF[(r0 + i) * 64 + c]);
      *(v8bf*)&sZb[b_idx(r0, c)] = o;
    }
  }
  __syncthreads();
  copy4096(zmat + (size_t)bh * 4096, sZa, t);
}

// ---------------------------------------------------------------------------
// Kernel 4/7: flash-style ker3 @ V partials (online softmax over N, 16 splits)
// ---------------------------------------------------------------------------
__global__ void __launch_bounds__(128)
kv_partial(const __bf16* __restrict__ qlandA, const __bf16* __restrict__ kb,
           const __bf16* __restrict__ vb, float* __restrict__ pacc,
           float* __restrict__ pm, float* __restrict__ pl) {
  __shared__ alignas(32) __bf16 sQf[4096];
  __shared__ alignas(32) __bf16 sKTb[4096];
  __shared__ alignas(32) __bf16 sVb[4096];
  __shared__ alignas(32) __bf16 sPf[4096];
  __shared__ alignas(16) float sS[4096];
  __shared__ float rm[64], rl[64], rfac[64];
  int bh = blockIdx.x / NSPLIT;
  int sp = blockIdx.x % NSPLIT;
  int t = threadIdx.x;
  int wave = t >> 5, lane = t & 31, cl = lane & 15, g = lane >> 4;
  copy4096(sQf, qlandA + (size_t)bh * 4096, t);
  if (t < 64) { rm[t] = -1e30f; rl[t] = 0.f; }
  v8f acc[4] = {};
  __syncthreads();
  v16bf qa0 = load_frag(sQf, wave * 2 + 0);
  v16bf qa1 = load_frag(sQf, wave * 2 + 1);
  int nbase = sp * (Nn / NSPLIT);
  for (int c = 0; c < NCHUNK; ++c) {
    int n0 = nbase + c * 64;
    const __bf16* kp = kb + ((size_t)bh * Nn + n0) * HD;
    const __bf16* vp = vb + ((size_t)bh * Nn + n0) * HD;
    for (int idx = t; idx < 512; idx += 128) {  // K^T: (k=d,n=tok), 8-wide in d
      int k8 = (idx & 7) << 3, n = idx >> 3;
      *(v8bf*)&sKTb[b_idx(k8, n)] = *(const v8bf*)&kp[(size_t)n * HD + k8];
    }
    for (int idx = t; idx < 512; idx += 128) {  // V: (k=tok,n=d), 16B gl loads
      int n8 = (idx & 7) << 3, k = idx >> 3;
      v8bf v = *(const v8bf*)&vp[(size_t)k * HD + n8];
#pragma unroll
      for (int i = 0; i < 8; ++i) sVb[b_idx(k, n8 + i)] = v[i];
    }
    if (c + 1 < NCHUNK) {
      __builtin_prefetch(kp + 4096 + t * 32, 0, 1);
      __builtin_prefetch(vp + 4096 + t * 32, 0, 1);
    }
    __syncthreads();
    {
      v16bf b0 = load_frag(sKTb, 0), b1 = load_frag(sKTb, 1);
#pragma unroll
      for (int ct = 0; ct < 4; ++ct) {
        v16bf n0 = b0, n1 = b1;
        if (ct < 3) {
          n0 = load_frag(sKTb, (ct + 1) * 2 + 0);
          n1 = load_frag(sKTb, (ct + 1) * 2 + 1);
        }
        v8f s = {};
        s = wmma_bf16(qa0, b0, s);
        s = wmma_bf16(qa1, b1, s);
        store_D_lds(sS, 64, wave * 16, ct * 16, s);
        b0 = n0; b1 = n1;
      }
    }
    __syncthreads();
    if (t < 64) {  // online softmax update for row t
      float mc = -1e30f;
      for (int j = 0; j < 64; j += 4) {
        v4f v = *(const v4f*)&sS[t * 64 + j];
        mc = fmaxf(mc, fmaxf(fmaxf(v[0], v[1]), fmaxf(v[2], v[3])));
      }
      float mnew = fmaxf(rm[t], mc);
      float fac = __expf(rm[t] - mnew);
      float sum = 0.f;
      for (int j8 = 0; j8 < 64; j8 += 8) {
        v8bf p8;
#pragma unroll
        for (int i = 0; i < 8; ++i) {
          float p = __expf(sS[t * 64 + j8 + i] - mnew);
          p8[i] = (__bf16)p;
          sum += p;
        }
        *(v8bf*)&sPf[a_idx(t, j8)] = p8;
      }
      rfac[t] = fac;
      rm[t] = mnew;
      rl[t] = rl[t] * fac + sum;
    }
    __syncthreads();
#pragma unroll
    for (int ct = 0; ct < 4; ++ct)
#pragma unroll
      for (int r = 0; r < 8; ++r) acc[ct][r] *= rfac[wave * 16 + r + 8 * g];
    {
      v16bf pa0 = load_frag(sPf, wave * 2 + 0);
      v16bf pa1 = load_frag(sPf, wave * 2 + 1);
      v16bf b0 = load_frag(sVb, 0), b1 = load_frag(sVb, 1);
#pragma unroll
      for (int ct = 0; ct < 4; ++ct) {
        v16bf n0 = b0, n1 = b1;
        if (ct < 3) {
          n0 = load_frag(sVb, (ct + 1) * 2 + 0);
          n1 = load_frag(sVb, (ct + 1) * 2 + 1);
        }
        acc[ct] = wmma_bf16(pa0, b0, acc[ct]);
        acc[ct] = wmma_bf16(pa1, b1, acc[ct]);
        b0 = n0; b1 = n1;
      }
    }
    __syncthreads();
  }
  float* pa = pacc + (size_t)blockIdx.x * 4096;
#pragma unroll
  for (int ct = 0; ct < 4; ++ct)
#pragma unroll
    for (int r = 0; r < 8; ++r)
      pa[(wave * 16 + r + 8 * g) * 64 + ct * 16 + cl] = acc[ct][r];
  if (t < 64) {
    pm[(size_t)blockIdx.x * 64 + t] = rm[t];
    pl[(size_t)blockIdx.x * 64 + t] = rl[t];
  }
}

// ---------------------------------------------------------------------------
// Kernel 5/7: combine flash partials -> kv; W = Z @ kv.
// Partial tiles streamed via GLOBAL_LOAD_ASYNC_TO_LDS_B128, double-buffered:
// issue slice s+1, s_wait_asynccnt 8 (slice s landed; async loads complete
// in order), consume, barrier.
// ---------------------------------------------------------------------------
__global__ void __launch_bounds__(128)
kv_combine(const float* __restrict__ pacc, const float* __restrict__ pm,
           const float* __restrict__ pl, const __bf16* __restrict__ zmat,
           __bf16* __restrict__ wmat) {
  __shared__ alignas(32) __bf16 sZa[4096];
  __shared__ alignas(32) __bf16 sKVb[4096];
  __shared__ alignas(16) float sF[4096];
  __shared__ alignas(16) float sBuf[2][4096];  // async landing buffers
  __shared__ float sE[NSPLIT * 64];
  __shared__ float sL[64];
  int bh = blockIdx.x;
  int t = threadIdx.x;
  if (t < 64) {
    float M = -1e30f;
    for (int s = 0; s < NSPLIT; ++s)
      M = fmaxf(M, pm[(size_t)(bh * NSPLIT + s) * 64 + t]);
    float L = 0.f;
    for (int s = 0; s < NSPLIT; ++s) {
      float f = __expf(pm[(size_t)(bh * NSPLIT + s) * 64 + t] - M);
      sE[s * 64 + t] = f;
      L += pl[(size_t)(bh * NSPLIT + s) * 64 + t] * f;
    }
    sL[t] = L;
  }
  copy4096(sZa, zmat + (size_t)bh * 4096, t);

  auto issue_slice = [&](int s, int buf) {
    const float* src = pacc + ((size_t)(bh * NSPLIT + s)) * 4096;
    float* dst = sBuf[buf];
    for (int i = t; i < 1024; i += 128) {  // 8 x b128 per thread
      unsigned lds = (unsigned)(uintptr_t)(dst + i * 4);
      const float* gp = src + i * 4;
      asm volatile("global_load_async_to_lds_b128 %0, %1, off"
                   :: "v"(lds), "v"(gp) : "memory");
    }
  };

  float accv[32];
#pragma unroll
  for (int j = 0; j < 32; ++j) accv[j] = 0.f;
  issue_slice(0, 0);
  for (int s = 0; s < NSPLIT; ++s) {
    if (s + 1 < NSPLIT) {
      issue_slice(s + 1, (s + 1) & 1);
      asm volatile("s_wait_asynccnt 0x8" ::: "memory");
    } else {
      asm volatile("s_wait_asynccnt 0x0" ::: "memory");
    }
    __syncthreads();
    float f = sE[s * 64 + (t >> 1)];  // this thread owns half of row t>>1
    const float* buf = sBuf[s & 1] + t * 32;
#pragma unroll
    for (int j = 0; j < 32; j += 4) {
      v4f v = *(const v4f*)&buf[j];
      accv[j + 0] += v[0] * f;
      accv[j + 1] += v[1] * f;
      accv[j + 2] += v[2] * f;
      accv[j + 3] += v[3] * f;
    }
    __syncthreads();
  }
  {
    int r = t >> 1, c0 = (t & 1) * 32;
    float L = sL[r];
    for (int j = 0; j < 32; ++j)
      sKVb[b_idx(r, c0 + j)] = (__bf16)(accv[j] / L);
  }
  __syncthreads();
  mm64_frag(sF, sZa, sKVb);  // W = Z @ kv
  __syncthreads();
  for (int idx = t; idx < 512; idx += 128) {  // wmat (B layout), 16B stores
    int c = idx & 63, r0 = (idx >> 6) << 3;
    v8bf o;
#pragma unroll
    for (int i = 0; i < 8; ++i) o[i] = (__bf16)sF[(r0 + i) * 64 + c];
    *(v8bf*)&wmat[(size_t)bh * 4096 + b_idx(r0, c)] = o;
  }
}

// ---------------------------------------------------------------------------
// Kernel 6/7: out_head = softmax(q @ k_land^T) @ W, 128-row q tile.
// ---------------------------------------------------------------------------
__global__ void __launch_bounds__(128)
ker1_apply(const __bf16* __restrict__ qb, const __bf16* __restrict__ klandB,
           const __bf16* __restrict__ wmatB, __bf16* __restrict__ attn) {
  __shared__ alignas(32) __bf16 sQf[8192];
  __shared__ alignas(32) __bf16 sKLb[4096];
  __shared__ alignas(32) __bf16 sWb[4096];
  __shared__ alignas(32) __bf16 sPf[8192];
  __shared__ alignas(16) float sS[8192];
  int bh = blockIdx.x >> 7;
  int n0 = (blockIdx.x & 127) * 128;
  int t = threadIdx.x;
  int wave = t >> 5, lane = t & 31, cl = lane & 15, g = lane >> 4;
  int b = bh >> 3, h = bh & 7;
  for (int idx = t; idx < 1024; idx += 128) {  // q tile, 8-wide in k
    int k8 = (idx & 7) << 3, m = idx >> 3;
    *(v8bf*)&sQf[a_idx(m, k8)] =
        *(const v8bf*)&qb[((size_t)bh * Nn + n0 + m) * HD + k8];
  }
  copy4096(sKLb, klandB + (size_t)bh * 4096, t);
  copy4096(sWb, wmatB + (size_t)bh * 4096, t);
  __syncthreads();
  {
    v16bf qa[2][2];
#pragma unroll
    for (int s = 0; s < 2; ++s) {
      qa[s][0] = load_frag(sQf, (wave * 2 + s) * 2 + 0);
      qa[s][1] = load_frag(sQf, (wave * 2 + s) * 2 + 1);
    }
    v16bf b0 = load_frag(sKLb, 0), b1 = load_frag(sKLb, 1);
#pragma unroll
    for (int ct = 0; ct < 4; ++ct) {  // B fragment feeds both strips
      v16bf nb0 = b0, nb1 = b1;
      if (ct < 3) {
        nb0 = load_frag(sKLb, (ct + 1) * 2 + 0);
        nb1 = load_frag(sKLb, (ct + 1) * 2 + 1);
      }
#pragma unroll
      for (int s = 0; s < 2; ++s) {
        v8f sc = {};
        sc = wmma_bf16(qa[s][0], b0, sc);
        sc = wmma_bf16(qa[s][1], b1, sc);
        store_D_lds(sS, 64, (wave * 2 + s) * 16, ct * 16, sc);
      }
      b0 = nb0; b1 = nb1;
    }
  }
  __syncthreads();
  {  // full-row softmax (m = 64), one row per thread, 8-wide stores
    float mx = -1e30f;
    for (int j = 0; j < 64; j += 4) {
      v4f v = *(const v4f*)&sS[t * 64 + j];
      mx = fmaxf(mx, fmaxf(fmaxf(v[0], v[1]), fmaxf(v[2], v[3])));
    }
    float sum = 0.f;
    for (int j = 0; j < 64; ++j) sum += __expf(sS[t * 64 + j] - mx);
    float inv = 1.f / sum;
    for (int j8 = 0; j8 < 64; j8 += 8) {
      v8bf p;
#pragma unroll
      for (int i = 0; i < 8; ++i)
        p[i] = (__bf16)(__expf(sS[t * 64 + j8 + i] - mx) * inv);
      *(v8bf*)&sPf[a_idx(t, j8)] = p;
    }
  }
  __syncthreads();
  {
    v16bf pa[2][2];
#pragma unroll
    for (int s = 0; s < 2; ++s) {
      pa[s][0] = load_frag(sPf, (wave * 2 + s) * 2 + 0);
      pa[s][1] = load_frag(sPf, (wave * 2 + s) * 2 + 1);
    }
    v16bf b0 = load_frag(sWb, 0), b1 = load_frag(sWb, 1);
#pragma unroll
    for (int ct = 0; ct < 4; ++ct) {
      v16bf nb0 = b0, nb1 = b1;
      if (ct < 3) {
        nb0 = load_frag(sWb, (ct + 1) * 2 + 0);
        nb1 = load_frag(sWb, (ct + 1) * 2 + 1);
      }
#pragma unroll
      for (int s = 0; s < 2; ++s) {
        v8f o = {};
        o = wmma_bf16(pa[s][0], b0, o);
        o = wmma_bf16(pa[s][1], b1, o);
#pragma unroll
        for (int r = 0; r < 8; ++r) {
          int lrow = (wave * 2 + s) * 16 + r + 8 * g;
          attn[((size_t)(b * Nn + n0 + lrow)) * Cc + h * 64 + ct * 16 + cl] =
              (__bf16)o[r];
        }
      }
      b0 = nb0; b1 = nb1;
    }
  }
}

// ---------------------------------------------------------------------------
// Kernel 7/7: final projection out = attn @ wo^T + bo, 128x64 tiles, f32 out.
// ---------------------------------------------------------------------------
__global__ void __launch_bounds__(128)
gemm_out(const __bf16* __restrict__ attn, const float* __restrict__ wo,
         const float* __restrict__ bo, float* __restrict__ out) {
  __shared__ alignas(32) __bf16 sAf[8192];
  __shared__ alignas(32) __bf16 sBf[4096];
  int t = threadIdx.x;
  int wave = t >> 5, lane = t & 31, cl = lane & 15, g = lane >> 4;
  int row0 = blockIdx.x * 128;
  int col0 = blockIdx.y * 64;
  v8f acc[2][4] = {};
  for (int kc = 0; kc < 8; ++kc) {
    int k0 = kc * 64;
    for (int idx = t; idx < 1024; idx += 128) {  // A tile, 16B global loads
      int k8 = (idx & 7) << 3, m = idx >> 3;
      *(v8bf*)&sAf[a_idx(m, k8)] =
          *(const v8bf*)&attn[(size_t)(row0 + m) * Cc + k0 + k8];
    }
    for (int idx = t; idx < 512; idx += 128) {
      int k8 = (idx & 7) << 3, n = idx >> 3;
      *(v8bf*)&sBf[b_idx(k8, n)] = cvt8(&wo[(size_t)(col0 + n) * Cc + k0 + k8]);
    }
    if (kc + 1 < 8)
      __builtin_prefetch(&attn[(size_t)(row0 + t) * Cc + k0 + 64], 0, 1);
    __syncthreads();
    v16bf a[2][2];
#pragma unroll
    for (int s = 0; s < 2; ++s) {
      a[s][0] = load_frag(sAf, (wave * 2 + s) * 2 + 0);
      a[s][1] = load_frag(sAf, (wave * 2 + s) * 2 + 1);
    }
    v16bf b0 = load_frag(sBf, 0), b1 = load_frag(sBf, 1);
#pragma unroll
    for (int ct = 0; ct < 4; ++ct) {
      v16bf nb0 = b0, nb1 = b1;
      if (ct < 3) {
        nb0 = load_frag(sBf, (ct + 1) * 2 + 0);
        nb1 = load_frag(sBf, (ct + 1) * 2 + 1);
      }
#pragma unroll
      for (int s = 0; s < 2; ++s) {
        acc[s][ct] = wmma_bf16(a[s][0], b0, acc[s][ct]);
        acc[s][ct] = wmma_bf16(a[s][1], b1, acc[s][ct]);
      }
      b0 = nb0; b1 = nb1;
    }
    __syncthreads();
  }
#pragma unroll
  for (int s = 0; s < 2; ++s)
#pragma unroll
    for (int ct = 0; ct < 4; ++ct)
#pragma unroll
      for (int r = 0; r < 8; ++r) {
        int gr = row0 + (wave * 2 + s) * 16 + r + 8 * g;
        int gc = col0 + ct * 16 + cl;
        out[(size_t)gr * Cc + gc] = acc[s][ct][r] + bo[gc];
      }
}

// ---------------------------------------------------------------------------
// Host side
// ---------------------------------------------------------------------------
extern "C" void kernel_launch(void* const* d_in, const int* in_sizes, int n_in,
                              void* d_out, int out_size, void* d_ws,
                              size_t ws_size, hipStream_t stream) {
  (void)in_sizes; (void)n_in; (void)out_size; (void)ws_size;
  const float* x  = (const float*)d_in[0];
  const float* wq = (const float*)d_in[1];
  const float* bq = (const float*)d_in[2];
  const float* wk = (const float*)d_in[3];
  const float* bk = (const float*)d_in[4];
  const float* wv = (const float*)d_in[5];
  const float* bv = (const float*)d_in[6];
  const float* wo = (const float*)d_in[7];
  const float* bo = (const float*)d_in[8];
  float* out = (float*)d_out;

  char* ws = (char*)d_ws;
  size_t off = 0;
  auto take = [&](size_t bytes) {
    char* p = ws + off;
    off += (bytes + 255) & ~(size_t)255;
    return p;
  };
  __bf16* qb     = (__bf16*)take((size_t)BH * Nn * HD * 2);
  __bf16* kb     = (__bf16*)take((size_t)BH * Nn * HD * 2);
  __bf16* vb     = (__bf16*)take((size_t)BH * Nn * HD * 2);
  __bf16* attn   = (__bf16*)take((size_t)Bb * Nn * Cc * 2);
  __bf16* qlandA = (__bf16*)take((size_t)BH * Mm * HD * 2);
  __bf16* klandB = (__bf16*)take((size_t)BH * Mm * HD * 2);
  __bf16* zmat   = (__bf16*)take((size_t)BH * Mm * Mm * 2);
  __bf16* wmat   = (__bf16*)take((size_t)BH * Mm * HD * 2);
  float* pacc    = (float*)take((size_t)BH * NSPLIT * Mm * HD * 4);
  float* pm      = (float*)take((size_t)BH * NSPLIT * Mm * 4);
  float* pl      = (float*)take((size_t)BH * NSPLIT * Mm * 4);

  gemm_qkv<<<dim3(Bb * Nn / 64, Cc / 64), 128, 0, stream>>>(
      x, wq, bq, wk, bk, wv, bv, qb, kb, vb);
  landmarks<<<BH * Mm, 64, 0, stream>>>(qb, kb, qlandA, klandB);
  ker2_pinv<<<BH, 128, 0, stream>>>(qlandA, klandB, zmat);
  kv_partial<<<BH * NSPLIT, 128, 0, stream>>>(qlandA, kb, vb, pacc, pm, pl);
  kv_combine<<<BH, 128, 0, stream>>>(pacc, pm, pl, zmat, wmat);
  ker1_apply<<<BH * (Nn / 128), 128, 0, stream>>>(qb, klandB, wmat, attn);
  gemm_out<<<dim3(Bb * Nn / 128, Cc / 64), 128, 0, stream>>>(attn, wo, bo, out);
}
